// SetAbstraction_1211180777511
// MI455X (gfx1250) — matrix operand order repaired and checked by reading, not compiled, not run on hardware
//
#include <hip/hip_runtime.h>
#include <hip/hip_bf16.h>
#include <stdint.h>

typedef __bf16 v16bf __attribute__((ext_vector_type(16)));
typedef __bf16 v8bf  __attribute__((ext_vector_type(8)));
typedef float  v8f   __attribute__((ext_vector_type(8)));

#define B_      16
#define N_      4096
#define S_      1024
#define K_      32
#define CINF_   64
#define CENTERS (B_ * S_)          // 16384
#define GEMM_WAVES 4
#define GEMM_BLOCKS (CENTERS / GEMM_WAVES)   // 4096
#define BN_COUNT ((float)(CENTERS * K_))     // 524288

// ---------------------------------------------------------------------------
// Kernel 1: gather sampled centers -> d_out[0 .. CENTERS*3)
// ---------------------------------------------------------------------------
__global__ __launch_bounds__(256) void gather_centers(
    const float* __restrict__ xyz, const int* __restrict__ fps,
    float* __restrict__ out) {
  int i = blockIdx.x * blockDim.x + threadIdx.x;
  if (i >= CENTERS * 3) return;
  int c = i / 3, d = i - c * 3;
  int b = c / S_, s = c - b * S_;
  out[i] = xyz[((size_t)b * N_ + fps[b * S_ + s]) * 3 + d];
}

// ---------------------------------------------------------------------------
// Wave32 bitonic helpers (shfl_xor based, deterministic (dist,idx) ordering)
// ---------------------------------------------------------------------------
__device__ __forceinline__ void cswap(float& v, int& i, int lane, int j, bool asc) {
  float ov = __shfl_xor(v, j);
  int   oi = __shfl_xor(i, j);
  bool lower   = (lane & j) == 0;
  bool takeMin = (lower == asc);
  bool partnerSmaller = (ov < v) || (ov == v && oi < i);
  if (partnerSmaller == takeMin) { v = ov; i = oi; }
}

__device__ __forceinline__ void bitonic_sort32(float& v, int& i, int lane) {
#pragma unroll
  for (int k = 2; k <= 32; k <<= 1) {
    const bool asc = (lane & k) == 0;
#pragma unroll
    for (int j = k >> 1; j > 0; j >>= 1) cswap(v, i, lane, j, asc);
  }
}

__device__ __forceinline__ void bitonic_merge32(float& v, int& i, int lane) {
#pragma unroll
  for (int j = 16; j > 0; j >>= 1) cswap(v, i, lane, j, true);
}

// ---------------------------------------------------------------------------
// Kernel 2: kNN. One wave (32 lanes) per center; resident top-32, one per lane.
// ---------------------------------------------------------------------------
__global__ __launch_bounds__(256) void knn_kernel(
    const float* __restrict__ xyz, const float* __restrict__ ctr,
    int* __restrict__ knn) {
  const int lane   = threadIdx.x & 31;
  const int wave   = threadIdx.x >> 5;
  const int center = blockIdx.x * 8 + wave;
  const int b      = center >> 10;  // /S_
  const float cx = ctr[center * 3 + 0];
  const float cy = ctr[center * 3 + 1];
  const float cz = ctr[center * 3 + 2];
  const float* px = xyz + (size_t)b * N_ * 3;

  float rd; int ri;
  {
    const float* p = px + lane * 3;
    float dx = p[0] - cx, dy = p[1] - cy, dz = p[2] - cz;
    rd = dx * dx + dy * dy + dz * dz;
    ri = lane;
  }
  bitonic_sort32(rd, ri, lane);

  for (int base = 32; base < N_; base += 32) {
    const int pi = base + lane;
    const float* p = px + (size_t)pi * 3;
    float dx = p[0] - cx, dy = p[1] - cy, dz = p[2] - cz;
    float d = dx * dx + dy * dy + dz * dz;
    float worst = __shfl(rd, 31);
    if (__ballot(d < worst) != 0ull) {      // wave-uniform skip
      int di = pi;
      bitonic_sort32(d, di, lane);
      // merge two sorted 32-lists: min(res[i], new[31-i]) then clean-up merge
      float rv  = __shfl(d, 31 - lane);
      int   rvi = __shfl(di, 31 - lane);
      if (rv < rd || (rv == rd && rvi < ri)) { rd = rv; ri = rvi; }
      bitonic_merge32(rd, ri, lane);
    }
  }
  knn[center * K_ + lane] = ri;
}

// ---------------------------------------------------------------------------
// Kernel 3/5/7: fused [gather|BN+ReLU] -> bf16 WMMA GEMM -> bias -> f32 store
//               + deterministic per-block channel sum / sumsq partials.
// One wave per center; 4 waves (128 threads) per block.
// yout layout: [center][neighbor(32)][COUT] f32 (channel-contiguous).
// ---------------------------------------------------------------------------
template <int CIN_REAL, int CIN_P, int COUT, bool FIRST>
__global__ __launch_bounds__(128) void fused_gemm_bn(
    const float* __restrict__ xyz, const float* __restrict__ feats,
    const float* __restrict__ ctr, const int* __restrict__ knn,
    const float* __restrict__ yprev, const float* __restrict__ ssprev,
    const float* __restrict__ W, const float* __restrict__ bias,
    float* __restrict__ yout, float* __restrict__ pb) {
  constexpr int MT = COUT / 16;   // M tiles
  constexpr int KC = CIN_P / 32;  // contraction chunks

  __shared__ __align__(32) __bf16 wsh[COUT * CIN_P];
  __shared__ __align__(32) __bf16 xsh[FIRST ? (GEMM_WAVES * K_ * CIN_P) : 8];
  __shared__ float sbias[COUT];
  __shared__ float sscale[FIRST ? 1 : CIN_REAL];
  __shared__ float sshift[FIRST ? 1 : CIN_REAL];
  __shared__ float swsum[GEMM_WAVES][COUT];
  __shared__ float swsq[GEMM_WAVES][COUT];

  const int tid  = threadIdx.x;
  const int wave = tid >> 5;
  const int lane = tid & 31;
  const int m16  = lane & 15;
  const int half = lane >> 4;
  const int center = blockIdx.x * GEMM_WAVES + wave;

  // ---- stage weights (f32 -> bf16, zero-padded), bias, BN scale/shift ----
  for (int i = tid; i < COUT * CIN_P; i += 128) {
    int o = i / CIN_P, c = i - o * CIN_P;
    float w = (c < CIN_REAL) ? W[o * CIN_REAL + c] : 0.0f;
    wsh[i] = (__bf16)w;
  }
  for (int i = tid; i < COUT; i += 128) sbias[i] = bias[i];
  if (!FIRST) {
    for (int i = tid; i < CIN_REAL; i += 128) {
      sscale[i] = ssprev[i];
      sshift[i] = ssprev[CIN_REAL + i];
    }
  }

  // ---- layer 0: gather grouped input into LDS (lane == neighbor) ----
  if (FIRST) {
    const int b  = center >> 10;
    const int nb = knn[center * K_ + lane];
    const float cx = ctr[center * 3 + 0];
    const float cy = ctr[center * 3 + 1];
    const float cz = ctr[center * 3 + 2];
    const float* p = xyz + ((size_t)b * N_ + nb) * 3;
    __bf16* xr = xsh + wave * (K_ * CIN_P) + lane * CIN_P;
    xr[0] = (__bf16)(p[0] - cx);
    xr[1] = (__bf16)(p[1] - cy);
    xr[2] = (__bf16)(p[2] - cz);
    const float* f = feats + ((size_t)b * N_ + nb) * CINF_;
#pragma unroll
    for (int c = 0; c < CINF_; ++c) xr[3 + c] = (__bf16)f[c];
#pragma unroll
    for (int c = 3 + CINF_; c < CIN_P; ++c) xr[c] = (__bf16)0.0f;
  }
  __syncthreads();

  // ---- WMMA main loop: D(16x16 f32) += A(16x32 bf16) * B(32x16 bf16) ----
  v8f acc[MT][2] = {};
#pragma unroll
  for (int kc = 0; kc < KC; ++kc) {
    // B fragments: lane holds column n, 16 contiguous contraction values
    v16bf bf[2];
#pragma unroll
    for (int tn = 0; tn < 2; ++tn) {
      const int n  = tn * 16 + m16;
      const int kb = kc * 32 + half * 16;
      if (FIRST) {
        bf[tn] = *reinterpret_cast<const v16bf*>(
            xsh + wave * (K_ * CIN_P) + n * CIN_P + kb);
      } else {
        const float* yp =
            yprev + (size_t)center * (K_ * CIN_REAL) + (size_t)n * CIN_REAL + kb;
        v16bf t;
#pragma unroll
        for (int e = 0; e < 16; ++e) {
          float v = fmaf(yp[e], sscale[kb + e], sshift[kb + e]);
          t[e] = (__bf16)fmaxf(v, 0.0f);
        }
        bf[tn] = t;
      }
    }
    // A fragments from LDS weights: 16 contiguous contraction values per lane
#pragma unroll
    for (int tm = 0; tm < MT; ++tm) {
      const __bf16* wr = wsh + (tm * 16 + m16) * CIN_P + kc * 32 + half * 16;
      v8bf lo = *reinterpret_cast<const v8bf*>(wr);
      v8bf hi = *reinterpret_cast<const v8bf*>(wr + 8);
      v16bf af = __builtin_shufflevector(lo, hi, 0, 1, 2, 3, 4, 5, 6, 7, 8, 9,
                                         10, 11, 12, 13, 14, 15);
      acc[tm][0] = __builtin_amdgcn_wmma_f32_16x16x32_bf16(
          false, af, false, bf[0], (short)0, acc[tm][0], false, false);
      acc[tm][1] = __builtin_amdgcn_wmma_f32_16x16x32_bf16(
          false, af, false, bf[1], (short)0, acc[tm][1], false, false);
    }
  }

  // ---- epilogue: +bias, store f32, per-channel sum/sumsq partials ----
  const size_t ybase = (size_t)center * (K_ * COUT);
#pragma unroll
  for (int tm = 0; tm < MT; ++tm) {
    float s8[8], q8[8];
#pragma unroll
    for (int r = 0; r < 8; ++r) { s8[r] = 0.0f; q8[r] = 0.0f; }
#pragma unroll
    for (int tn = 0; tn < 2; ++tn) {
      const int n = tn * 16 + m16;
      float vals[8];
#pragma unroll
      for (int r = 0; r < 8; ++r) {
        const int o = tm * 16 + half * 8 + r;
        float v = acc[tm][tn][r] + sbias[o];
        vals[r] = v;
        s8[r] += v;
        q8[r] += v * v;
      }
      float* dst = yout + ybase + (size_t)n * COUT + tm * 16 + half * 8;
      *(float4*)(dst + 0) = make_float4(vals[0], vals[1], vals[2], vals[3]);
      *(float4*)(dst + 4) = make_float4(vals[4], vals[5], vals[6], vals[7]);
    }
    // reduce across the 16 columns held by this half-wave
#pragma unroll
    for (int r = 0; r < 8; ++r) {
      float s = s8[r], q = q8[r];
#pragma unroll
      for (int m = 1; m <= 8; m <<= 1) {
        s += __shfl_xor(s, m);
        q += __shfl_xor(q, m);
      }
      if (m16 == 0) {  // lanes 0 and 16
        const int o = tm * 16 + half * 8 + r;
        swsum[wave][o] = s;
        swsq[wave][o]  = q;
      }
    }
  }
  __syncthreads();
  // fixed-order block partials (deterministic, no float atomics)
  for (int i = tid; i < COUT; i += 128) {
    float s = 0.0f, q = 0.0f;
#pragma unroll
    for (int w = 0; w < GEMM_WAVES; ++w) { s += swsum[w][i]; q += swsq[w][i]; }
    pb[(size_t)blockIdx.x * 2 * COUT + i]        = s;
    pb[(size_t)blockIdx.x * 2 * COUT + COUT + i] = q;
  }
}

// ---------------------------------------------------------------------------
// Kernel 4/6/8: fold BN stats into fused scale/shift (deterministic order)
// ---------------------------------------------------------------------------
template <int COUT>
__global__ void finalize_bn(const float* __restrict__ pb,
                            const float* __restrict__ g,
                            const float* __restrict__ bt,
                            float* __restrict__ ss) {
  int o = threadIdx.x;
  if (o >= COUT) return;
  float s = 0.0f, q = 0.0f;
  for (int bk = 0; bk < GEMM_BLOCKS; ++bk) {
    s += pb[(size_t)bk * 2 * COUT + o];
    q += pb[(size_t)bk * 2 * COUT + COUT + o];
  }
  const float inv = 1.0f / BN_COUNT;
  float mu  = s * inv;
  float var = q * inv - mu * mu;
  float rs  = rsqrtf(var + 1e-5f);
  float sc  = g[o] * rs;
  ss[o]        = sc;
  ss[COUT + o] = bt[o] - mu * sc;
}

// ---------------------------------------------------------------------------
// Kernel 9: BN+ReLU on layer-2 pre-activations + max over K neighbors
// ---------------------------------------------------------------------------
__global__ __launch_bounds__(256) void bn_relu_maxpool(
    const float* __restrict__ y2, const float* __restrict__ ss,
    float* __restrict__ out) {
  int gid = blockIdx.x * blockDim.x + threadIdx.x;
  if (gid >= CENTERS * 128) return;
  int center = gid >> 7, o = gid & 127;
  float sc = ss[o], sh = ss[128 + o];
  const float* yp = y2 + (size_t)center * (K_ * 128) + o;
  float m = -INFINITY;
#pragma unroll 4
  for (int k = 0; k < K_; ++k) {
    float v = fmaxf(fmaf(yp[(size_t)k * 128], sc, sh), 0.0f);
    m = fmaxf(m, v);
  }
  out[gid] = m;
}

// ---------------------------------------------------------------------------
extern "C" void kernel_launch(void* const* d_in, const int* in_sizes, int n_in,
                              void* d_out, int out_size, void* d_ws,
                              size_t ws_size, hipStream_t stream) {
  (void)in_sizes; (void)n_in; (void)out_size; (void)ws_size;
  const float* xyz   = (const float*)d_in[0];
  const float* feats = (const float*)d_in[1];
  const int*   fps   = (const int*)d_in[2];
  const float* W0  = (const float*)d_in[3];
  const float* b0  = (const float*)d_in[4];
  const float* g0  = (const float*)d_in[5];
  const float* bt0 = (const float*)d_in[6];
  const float* W1  = (const float*)d_in[7];
  const float* b1  = (const float*)d_in[8];
  const float* g1  = (const float*)d_in[9];
  const float* bt1 = (const float*)d_in[10];
  const float* W2  = (const float*)d_in[11];
  const float* b2  = (const float*)d_in[12];
  const float* g2  = (const float*)d_in[13];
  const float* bt2 = (const float*)d_in[14];
  float* out = (float*)d_out;

  // workspace carving
  uint8_t* w = (uint8_t*)d_ws;
  int* knn = (int*)w;        w += (size_t)CENTERS * K_ * sizeof(int);     // 2 MiB
  float* yA = (float*)w;     w += (size_t)CENTERS * K_ * 128 * 4;         // 256 MiB (y0, then y2)
  float* y1 = (float*)w;     w += (size_t)CENTERS * K_ * 64 * 4;          // 128 MiB
  float* pb = (float*)w;     w += (size_t)GEMM_BLOCKS * 2 * 128 * 4;      // 4 MiB (reused per layer)
  float* ss0 = (float*)w;    w += 128 * 4;
  float* ss1 = (float*)w;    w += 128 * 4;
  float* ss2 = (float*)w;    w += 256 * 4;

  gather_centers<<<(CENTERS * 3 + 255) / 256, 256, 0, stream>>>(xyz, fps, out);
  knn_kernel<<<CENTERS / 8, 256, 0, stream>>>(xyz, out, knn);

  fused_gemm_bn<67, 96, 64, true><<<GEMM_BLOCKS, 128, 0, stream>>>(
      xyz, feats, out, knn, nullptr, nullptr, W0, b0, yA, pb);
  finalize_bn<64><<<1, 64, 0, stream>>>(pb, g0, bt0, ss0);

  fused_gemm_bn<64, 64, 64, false><<<GEMM_BLOCKS, 128, 0, stream>>>(
      nullptr, nullptr, nullptr, nullptr, yA, ss0, W1, b1, y1, pb);
  finalize_bn<64><<<1, 64, 0, stream>>>(pb, g1, bt1, ss1);

  fused_gemm_bn<64, 64, 128, false><<<GEMM_BLOCKS, 128, 0, stream>>>(
      nullptr, nullptr, nullptr, nullptr, y1, ss1, W2, b2, yA, pb);
  finalize_bn<128><<<1, 128, 0, stream>>>(pb, g2, bt2, ss2);

  bn_relu_maxpool<<<(CENTERS * 128) / 256, 256, 0, stream>>>(yA, ss2,
                                                             out + CENTERS * 3);
}
